// DonutSwinLayer_28295244546145
// MI455X (gfx1250) — matrix operand order repaired and checked by reading, not compiled, not run on hardware
//
#include <hip/hip_runtime.h>
#include <hip/hip_bf16.h>

// ---------------- static problem config (matches reference) ----------------
#define BB   8
#define HH   80
#define WW   60
#define CC   512
#define WS_  10
#define SHIFT_ 5
#define NH_  16
#define HD_  32
#define LL   100          // tokens per window
#define NWIN 48           // windows per image
#define NWG  (BB * NWIN)  // 384 total windows
#define MTOK (BB * HH * WW)   // 38400 rows
#define LPAD 112          // L padded to mult of 16
#define KPAD 128          // L padded to mult of 32 (attn @ V inner dim)
#define C4   (4 * CC)     // 2048

typedef __attribute__((ext_vector_type(16))) _Float16 v16h;
typedef __attribute__((ext_vector_type(8)))  _Float16 v8h;
typedef __attribute__((ext_vector_type(8)))  float    v8f;

// ---------------- WMMA fragment helpers (contiguous k order, same in A and B) ----------
// A fragment 16x32 f16: lane<16 -> row=lane, K k0..k0+15 (contiguous)
//                        lane>=16 -> row=lane-16, K k0+16..k0+31
// caller passes p already offset by (row*lda + k0 + half*16)
static __device__ inline v16h ld_frag_a(const _Float16* __restrict__ p) {
  v8h lo = *(const v8h*)(p);
  v8h hi = *(const v8h*)(p + 8);
  v16h r;
#pragma unroll
  for (int i = 0; i < 8; ++i) { r[i] = lo[i]; r[i + 8] = hi[i]; }
  return r;
}
// B fragment 32x16 f16 fed from Bt (N x K row-major): lane holds column
// n=lane&15, K values k0 + (lane<16?0:16) + 0..15 (contiguous in Bt row)
static __device__ inline v16h ld_frag_b(const _Float16* __restrict__ p) {
  v8h lo = *(const v8h*)(p);
  v8h hi = *(const v8h*)(p + 8);
  v16h r;
#pragma unroll
  for (int i = 0; i < 8; ++i) { r[i] = lo[i]; r[i + 8] = hi[i]; }
  return r;
}
static __device__ inline v8f wmma16x16x32(v16h a, v16h b, v8f c) {
  return __builtin_amdgcn_wmma_f32_16x16x32_f16(false, a, false, b, (short)0, c,
                                                false, false);
}
static __device__ inline v8f zero8() {
  v8f z = {0.f, 0.f, 0.f, 0.f, 0.f, 0.f, 0.f, 0.f};
  return z;
}

// ---------------- small utility kernels ------------------------------------
__global__ __launch_bounds__(256) void k_transpose_f32_to_f16(
    const float* __restrict__ src, _Float16* __restrict__ dst, int R, int C) {
  int i = blockIdx.x * 256 + threadIdx.x;
  if (i >= R * C) return;
  int r = i / C, c = i - r * C;
  dst[(size_t)c * R + r] = (_Float16)src[i];
}

__global__ __launch_bounds__(256) void k_zero_f16(_Float16* __restrict__ p,
                                                  long long n) {
  long long i = (long long)blockIdx.x * 256 + threadIdx.x;
  if (i < n) p[i] = (_Float16)0.f;
}

// LN1 + cyclic shift + window partition -> Xw f16 (window-order rows), rowmap
__global__ __launch_bounds__(256) void k_ln_shift(
    const float* __restrict__ x, const float* __restrict__ g,
    const float* __restrict__ bta, _Float16* __restrict__ Xw,
    int* __restrict__ rowmap) {
  int r = blockIdx.x;                         // window-order row
  int b = r / (HH * WW); int rem = r - b * (HH * WW);
  int w = rem / LL;      int t = rem - w * LL;
  int wi = w / (WW / WS_), wj = w - wi * (WW / WS_);
  int ti = t / WS_,        tj = t - ti * WS_;
  int sh = wi * WS_ + ti, sw = wj * WS_ + tj;      // shifted-frame pixel
  int srh = sh + SHIFT_; if (srh >= HH) srh -= HH; // roll(-SHIFT) source
  int srw = sw + SHIFT_; if (srw >= WW) srw -= WW;
  int src = b * (HH * WW) + srh * WW + srw;

  const float* row = x + (size_t)src * CC;
  int tid = threadIdx.x;
  float x0 = row[tid], x1 = row[tid + 256];
  __shared__ float s1[256], s2[256];
  s1[tid] = x0 + x1; s2[tid] = x0 * x0 + x1 * x1;
  __syncthreads();
  for (int off = 128; off > 0; off >>= 1) {
    if (tid < off) { s1[tid] += s1[tid + off]; s2[tid] += s2[tid + off]; }
    __syncthreads();
  }
  float mu = s1[0] * (1.f / CC);
  float var = s2[0] * (1.f / CC) - mu * mu;
  float rs = rsqrtf(var + 1e-5f);
  _Float16* orow = Xw + (size_t)r * CC;
  orow[tid]       = (_Float16)((x0 - mu) * rs * g[tid] + bta[tid]);
  orow[tid + 256] = (_Float16)((x1 - mu) * rs * g[tid + 256] + bta[tid + 256]);
  if (tid == 0) rowmap[r] = src;
}

// LN2 over hs (token order) -> X2 f16
__global__ __launch_bounds__(256) void k_ln2(const float* __restrict__ hs,
                                             const float* __restrict__ g,
                                             const float* __restrict__ bta,
                                             _Float16* __restrict__ X2) {
  int r = blockIdx.x;
  const float* row = hs + (size_t)r * CC;
  int tid = threadIdx.x;
  float x0 = row[tid], x1 = row[tid + 256];
  __shared__ float s1[256], s2[256];
  s1[tid] = x0 + x1; s2[tid] = x0 * x0 + x1 * x1;
  __syncthreads();
  for (int off = 128; off > 0; off >>= 1) {
    if (tid < off) { s1[tid] += s1[tid + off]; s2[tid] += s2[tid + off]; }
    __syncthreads();
  }
  float mu = s1[0] * (1.f / CC);
  float var = s2[0] * (1.f / CC) - mu * mu;
  float rs = rsqrtf(var + 1e-5f);
  _Float16* orow = X2 + (size_t)r * CC;
  orow[tid]       = (_Float16)((x0 - mu) * rs * g[tid] + bta[tid]);
  orow[tid + 256] = (_Float16)((x1 - mu) * rs * g[tid + 256] + bta[tid + 256]);
}

// ---------------- generic WMMA GEMM: out = A(MxK) * Bt(NxK)^T + bias -------
enum { EP_F16 = 0, EP_VT = 1, EP_OUTPROJ = 2, EP_GELU = 3, EP_RESID = 4 };

template <int EP>
__global__ __launch_bounds__(256) void k_gemm(
    const _Float16* __restrict__ A, const _Float16* __restrict__ Bt,
    const float* __restrict__ bias, void* __restrict__ outp,
    const float* __restrict__ aux, const int* __restrict__ rowmap, int M,
    int N, int K, int lda, int ldb, int ldo) {
  const int wave = blockIdx.x * 8 + (threadIdx.x >> 5);
  const int nGroups = N >> 6;                 // each wave: 16 rows x 64 cols
  const int mTiles = M >> 4;
  if (wave >= mTiles * nGroups) return;
  const int mt = wave / nGroups, ng = wave - mt * nGroups;
  const int lane = threadIdx.x & 31, half = lane >> 4, l16 = lane & 15;
  const int mBase = mt << 4, nBase = ng << 6;

  v8f acc[4];
#pragma unroll
  for (int j = 0; j < 4; ++j) acc[j] = zero8();

  const _Float16* aRow = A + (size_t)(mBase + l16) * lda + half * 16;
  for (int k0 = 0; k0 < K; k0 += 32) {
    if (k0 + 32 < K) __builtin_prefetch(aRow + k0 + 32, 0, 1);
    v16h a = ld_frag_a(aRow + k0);
#pragma unroll
    for (int j = 0; j < 4; ++j) {
      const _Float16* bRow =
          Bt + (size_t)(nBase + j * 16 + l16) * ldb + k0 + half * 16;
      v16h b = ld_frag_b(bRow);
      acc[j] = wmma16x16x32(a, b, acc[j]);
    }
  }

#pragma unroll
  for (int j = 0; j < 4; ++j) {
    const int n = nBase + j * 16 + l16;
    const float bv = bias[n];
#pragma unroll
    for (int r = 0; r < 8; ++r) {
      const int m = mBase + r + half * 8;     // C/D layout (ISA 7.12.2)
      float v = acc[j][r] + bv;
      if (EP == EP_F16) {
        ((_Float16*)outp)[(size_t)m * ldo + n] = (_Float16)v;
      } else if (EP == EP_VT) {               // scatter V transposed per (w,h)
        int w = m / LL, t = m - w * LL;
        int h = n >> 5, d = n & 31;
        ((_Float16*)outp)[(((size_t)(w * NH_ + h) * HD_ + d) << 7) + t] =
            (_Float16)v;
      } else if (EP == EP_OUTPROJ) {          // window-reverse + residual
        int dst = rowmap[m];
        ((float*)outp)[(size_t)dst * ldo + n] =
            aux[(size_t)dst * ldo + n] + v;
      } else if (EP == EP_GELU) {
        float gg = 0.5f * v * (1.f + erff(v * 0.70710678118654752f));
        ((_Float16*)outp)[(size_t)m * ldo + n] = (_Float16)gg;
      } else if (EP == EP_RESID) {
        ((float*)outp)[(size_t)m * ldo + n] = aux[(size_t)m * ldo + n] + v;
      }
    }
  }
}

// ---------------- attention: scores = Q K^T / sqrt(HD) ---------------------
// one WMMA per 16x16 score tile (K = HD = 32 exactly)
__global__ __launch_bounds__(256) void k_scores(
    const _Float16* __restrict__ Q, const _Float16* __restrict__ Kf,
    float* __restrict__ scores) {
  int wave = blockIdx.x * 8 + (threadIdx.x >> 5);
  const int TOT = NWG * NH_ * 7 * 7;
  if (wave >= TOT) return;
  int nt = wave % 7; int tmp = wave / 7;
  int mt = tmp % 7;  tmp /= 7;
  int h = tmp % NH_; int w = tmp / NH_;
  int lane = threadIdx.x & 31, half = lane >> 4, l16 = lane & 15;

  const _Float16* ap =
      Q + (size_t)(w * LL + mt * 16 + l16) * CC + h * HD_ + half * 16;
  const _Float16* bp =
      Kf + (size_t)(w * LL + nt * 16 + l16) * CC + h * HD_ + half * 16;
  v16h a = ld_frag_a(ap);
  v16h b = ld_frag_b(bp);
  v8f c = zero8();
  c = wmma16x16x32(a, b, c);

  float* out = scores + ((size_t)(w * NH_ + h) * LPAD + mt * 16) * LPAD +
               nt * 16 + l16;
#pragma unroll
  for (int r = 0; r < 8; ++r)
    out[(size_t)(r + half * 8) * LPAD] = c[r] * 0.17677669529663687f;
}

// ---------------- softmax (+rel-pos bias +shift mask) -> attn f16 ----------
__global__ __launch_bounds__(128) void k_softmax(
    const float* __restrict__ scores, const float* __restrict__ relT,
    _Float16* __restrict__ attn) {
  int idx = blockIdx.x;                       // (w*NH + h)*LL + m
  int m = idx % LL; int wh = idx / LL;
  int h = wh % NH_; int w = wh / NH_;
  int n = threadIdx.x;

  float s = -1e30f;
  if (n < LL) {
    s = scores[((size_t)wh * LPAD + m) * LPAD + n];
    int im = m / WS_, jm = m - im * WS_;
    int in_ = n / WS_, jn = n - in_ * WS_;
    int ridx = (im - in_ + WS_ - 1) * (2 * WS_ - 1) + (jm - jn + WS_ - 1);
    s += relT[ridx * NH_ + h];
    // shift-window mask: region labels on partitioned (shifted) image
    int wl = w % NWIN;
    int wi = wl / (WW / WS_), wj = wl - wi * (WW / WS_);
    int phm = wi * WS_ + im, pwm = wj * WS_ + jm;
    int phn = wi * WS_ + in_, pwn = wj * WS_ + jn;
    int lm = (phm < HH - WS_ ? 0 : (phm < HH - SHIFT_ ? 1 : 2)) * 3 +
             (pwm < WW - WS_ ? 0 : (pwm < WW - SHIFT_ ? 1 : 2));
    int ln2 = (phn < HH - WS_ ? 0 : (phn < HH - SHIFT_ ? 1 : 2)) * 3 +
              (pwn < WW - WS_ ? 0 : (pwn < WW - SHIFT_ ? 1 : 2));
    if (lm != ln2) s -= 100.0f;
  }

  __shared__ float red[128];
  red[n] = s;
  __syncthreads();
  for (int off = 64; off > 0; off >>= 1) {
    if (n < off) red[n] = fmaxf(red[n], red[n + off]);
    __syncthreads();
  }
  float mx = red[0];
  __syncthreads();
  float e = (n < LL) ? __expf(s - mx) : 0.f;
  red[n] = e;
  __syncthreads();
  for (int off = 64; off > 0; off >>= 1) {
    if (n < off) red[n] += red[n + off];
    __syncthreads();
  }
  float inv = 1.f / red[0];
  // write all 128 cols: real cols get prob, pad cols (100..127) exact zero
  attn[((size_t)wh * LPAD + m) * KPAD + n] = (_Float16)(e * inv);
}

// ---------------- ctx = attn(100x128pad) @ V(128x32), via Vt ---------------
__global__ __launch_bounds__(256) void k_ctx(const _Float16* __restrict__ attn,
                                             const _Float16* __restrict__ Vt,
                                             _Float16* __restrict__ ctx) {
  int wave = blockIdx.x * 8 + (threadIdx.x >> 5);
  const int TOT = NWG * NH_ * 7 * 2;
  if (wave >= TOT) return;
  int nt = wave & 1; int tmp = wave >> 1;
  int mt = tmp % 7;  tmp /= 7;
  int h = tmp % NH_; int w = tmp / NH_;
  int lane = threadIdx.x & 31, half = lane >> 4, l16 = lane & 15;
  int wh = w * NH_ + h;

  const _Float16* aRow =
      attn + ((size_t)wh * LPAD + mt * 16 + l16) * KPAD + half * 16;
  const _Float16* bRow =
      Vt + ((size_t)wh * HD_ + nt * 16 + l16) * KPAD + half * 16;

  v8f c = zero8();
#pragma unroll
  for (int k0 = 0; k0 < KPAD; k0 += 32) {
    v16h a = ld_frag_a(aRow + k0);
    v16h b = ld_frag_b(bRow + k0);
    c = wmma16x16x32(a, b, c);
  }
#pragma unroll
  for (int r = 0; r < 8; ++r) {
    int m = mt * 16 + r + half * 8;
    if (m < LL) {
      int n = h * HD_ + nt * 16 + l16;
      ctx[(size_t)(w * LL + m) * CC + n] = (_Float16)c[r];
    }
  }
}

// ---------------- host-side orchestration ----------------------------------
extern "C" void kernel_launch(void* const* d_in, const int* in_sizes, int n_in,
                              void* d_out, int out_size, void* d_ws,
                              size_t ws_size, hipStream_t stream) {
  (void)in_sizes; (void)n_in; (void)out_size; (void)ws_size;
  const float* hidden = (const float*)d_in[0];
  const float* ln1g = (const float*)d_in[1];
  const float* ln1b = (const float*)d_in[2];
  const float* wq = (const float*)d_in[3];
  const float* bq = (const float*)d_in[4];
  const float* wk = (const float*)d_in[5];
  const float* bk = (const float*)d_in[6];
  const float* wv = (const float*)d_in[7];
  const float* bv = (const float*)d_in[8];
  const float* wo = (const float*)d_in[9];
  const float* bo = (const float*)d_in[10];
  const float* relb = (const float*)d_in[11];
  const float* ln2g = (const float*)d_in[12];
  const float* ln2b = (const float*)d_in[13];
  const float* w1 = (const float*)d_in[14];
  const float* b1 = (const float*)d_in[15];
  const float* w2 = (const float*)d_in[16];
  const float* b2 = (const float*)d_in[17];

  char* base = (char*)d_ws;
  size_t off = 0;
  auto alloc = [&](size_t bytes) -> void* {
    void* p = base + off;
    off = (off + bytes + 255) & ~(size_t)255;
    return p;
  };

  const int MSLACK = 16;  // extra rows so padded-tile reads stay in-bounds
  _Float16* WqT = (_Float16*)alloc((size_t)CC * CC * 2);
  _Float16* WkT = (_Float16*)alloc((size_t)CC * CC * 2);
  _Float16* WvT = (_Float16*)alloc((size_t)CC * CC * 2);
  _Float16* WoT = (_Float16*)alloc((size_t)CC * CC * 2);
  _Float16* W1T = (_Float16*)alloc((size_t)C4 * CC * 2);
  _Float16* W2T = (_Float16*)alloc((size_t)CC * C4 * 2);
  _Float16* Xw  = (_Float16*)alloc((size_t)MTOK * CC * 2);
  _Float16* Qf  = (_Float16*)alloc((size_t)(MTOK + MSLACK) * CC * 2);
  _Float16* Kf  = (_Float16*)alloc((size_t)(MTOK + MSLACK) * CC * 2);
  _Float16* Vt  = (_Float16*)alloc((size_t)NWG * NH_ * HD_ * KPAD * 2);
  float*    Sc  = (float*)alloc((size_t)NWG * NH_ * LPAD * LPAD * 4);
  _Float16* At  = (_Float16*)alloc((size_t)NWG * NH_ * LPAD * KPAD * 2);
  _Float16* Cx  = (_Float16*)alloc((size_t)MTOK * CC * 2);
  int*      rmap = (int*)alloc((size_t)MTOK * 4);
  float*    hs  = (float*)alloc((size_t)MTOK * CC * 4);
  _Float16* X2  = (_Float16*)alloc((size_t)MTOK * CC * 2);
  _Float16* Y1  = (_Float16*)alloc((size_t)MTOK * C4 * 2);
  float*    out = (float*)d_out;

  // 1) weight transpose+convert to f16  (Bt layout: N x K)
  int nw = CC * CC;
  k_transpose_f32_to_f16<<<(nw + 255) / 256, 256, 0, stream>>>(wq, WqT, CC, CC);
  k_transpose_f32_to_f16<<<(nw + 255) / 256, 256, 0, stream>>>(wk, WkT, CC, CC);
  k_transpose_f32_to_f16<<<(nw + 255) / 256, 256, 0, stream>>>(wv, WvT, CC, CC);
  k_transpose_f32_to_f16<<<(nw + 255) / 256, 256, 0, stream>>>(wo, WoT, CC, CC);
  int nw1 = CC * C4;
  k_transpose_f32_to_f16<<<(nw1 + 255) / 256, 256, 0, stream>>>(w1, W1T, CC, C4);
  k_transpose_f32_to_f16<<<(nw1 + 255) / 256, 256, 0, stream>>>(w2, W2T, C4, CC);

  // 2) LN1 + shift + window partition (and rowmap for the inverse scatter)
  k_ln_shift<<<MTOK, 256, 0, stream>>>(hidden, ln1g, ln1b, Xw, rmap);

  // 3) zero Vt (pad K-columns must be exact zero)
  long long vtN = (long long)NWG * NH_ * HD_ * KPAD;
  k_zero_f16<<<(unsigned)((vtN + 255) / 256), 256, 0, stream>>>(Vt, vtN);

  // 4) Q,K,V projections (f16-out WMMA GEMMs; V scattered transposed)
  const int gemmBlocks_qkv = (MTOK / 16) * (CC / 64) / 8;  // 2400
  k_gemm<EP_F16><<<gemmBlocks_qkv, 256, 0, stream>>>(
      Xw, WqT, bq, Qf, nullptr, nullptr, MTOK, CC, CC, CC, CC, CC);
  k_gemm<EP_F16><<<gemmBlocks_qkv, 256, 0, stream>>>(
      Xw, WkT, bk, Kf, nullptr, nullptr, MTOK, CC, CC, CC, CC, CC);
  k_gemm<EP_VT><<<gemmBlocks_qkv, 256, 0, stream>>>(
      Xw, WvT, bv, Vt, nullptr, nullptr, MTOK, CC, CC, CC, CC, 0);

  // 5) attention scores (one WMMA per tile, K == HD == 32)
  k_scores<<<(NWG * NH_ * 49) / 8, 256, 0, stream>>>(Qf, Kf, Sc);

  // 6) softmax with rel-pos bias + shift mask -> attn f16 (zero pad cols)
  k_softmax<<<NWG * NH_ * LL, 128, 0, stream>>>(Sc, relb, At);

  // 7) ctx = attn @ V
  k_ctx<<<(NWG * NH_ * 14) / 8, 256, 0, stream>>>(At, Vt, Cx);

  // 8) out-projection + window-reverse/unshift + residual -> hs (f32)
  k_gemm<EP_OUTPROJ><<<gemmBlocks_qkv, 256, 0, stream>>>(
      Cx, WoT, bo, hs, hidden, rmap, MTOK, CC, CC, CC, CC, CC);

  // 9) LN2
  k_ln2<<<MTOK, 256, 0, stream>>>(hs, ln2g, ln2b, X2);

  // 10) FFN: gelu(X2 @ W1 + b1) @ W2 + b2, + hs residual -> d_out
  const int gemmBlocks_ffn1 = (MTOK / 16) * (C4 / 64) / 8;  // 9600
  k_gemm<EP_GELU><<<gemmBlocks_ffn1, 256, 0, stream>>>(
      X2, W1T, b1, Y1, nullptr, nullptr, MTOK, C4, CC, CC, CC, C4);
  k_gemm<EP_RESID><<<gemmBlocks_qkv, 256, 0, stream>>>(
      Y1, W2T, b2, out, hs, nullptr, MTOK, CC, C4, C4, C4, CC);
}
